// SelfAttention_66666482368554
// MI455X (gfx1250) — compile-verified, matrix operand order checked by reading, not run on hardware
//
#include <hip/hip_runtime.h>

typedef __attribute__((ext_vector_type(16))) __bf16 v16bf;
typedef __attribute__((ext_vector_type(8)))  float  v8f;
typedef __attribute__((ext_vector_type(4)))  int    v4i;

#define CIN 256

static __device__ __forceinline__ v8f wmma_bf16(v16bf a, v16bf b, v8f c) {
    return __builtin_amdgcn_wmma_f32_16x16x32_bf16(
        /*neg_a=*/false, a, /*neg_b=*/false, b,
        /*c_mod=*/(short)0, c, /*reuse_a=*/false, /*reuse_b=*/false);
}

// ---- gfx1250 async global->LDS copy (ASYNCcnt) with sync fallback --------
#if __has_builtin(__builtin_amdgcn_global_load_async_to_lds_b128) && \
    __has_builtin(__builtin_amdgcn_s_wait_asynccnt)
#define USE_ASYNC_LDS 1
#else
#define USE_ASYNC_LDS 0
#endif

static __device__ __forceinline__ void copy16_g2l(const __bf16* g, __bf16* l) {
#if USE_ASYNC_LDS
    __builtin_amdgcn_global_load_async_to_lds_b128(
        (__attribute__((address_space(1))) v4i*)g,
        (__attribute__((address_space(3))) v4i*)l, 0, 0);
#else
    *(uint4*)l = *(const uint4*)g;
#endif
}

template <int N>
static __device__ __forceinline__ void async_wait() {
#if USE_ASYNC_LDS
    __builtin_amdgcn_s_wait_asynccnt((unsigned short)N);
#endif
}

// ---------------------------------------------------------------------------
// Spectral norm: one power iteration. sig = ||l2n(u@W^T) @ W||_2.
// Writes transposed bf16 weight: wt[out][in] = W[in][out] / sig.
// grid.x = 4 (theta, phi, g, attn)
// ---------------------------------------------------------------------------
__global__ __launch_bounds__(256) void sn_kernel(
    const float* w0, const float* u0, __bf16* o0,   // [256,32]
    const float* w1, const float* u1, __bf16* o1,   // [256,32]
    const float* w2, const float* u2, __bf16* o2,   // [256,128]
    const float* w3, const float* u3, __bf16* o3)   // [128,256]
{
    const float* w; const float* u; __bf16* o; int IN, OUT;
    switch (blockIdx.x) {
        case 0:  w = w0; u = u0; o = o0; IN = 256; OUT = 32;  break;
        case 1:  w = w1; u = u1; o = o1; IN = 256; OUT = 32;  break;
        case 2:  w = w2; u = u2; o = o2; IN = 256; OUT = 128; break;
        default: w = w3; u = u3; o = o3; IN = 128; OUT = 256; break;
    }
    __shared__ float v[256];
    __shared__ float red[256];
    __shared__ float nrm, isig;
    int t = threadIdx.x;

    float ti = 0.f;
    if (t < IN)
        for (int j = 0; j < OUT; ++j) ti += u[j] * w[t * OUT + j];
    red[t] = ti * ti;
    __syncthreads();
    for (int s = 128; s > 0; s >>= 1) { if (t < s) red[t] += red[t + s]; __syncthreads(); }
    if (t == 0) nrm = rsqrtf(red[0] + 1e-12f);
    __syncthreads();
    if (t < IN) v[t] = ti * nrm;
    __syncthreads();

    float so = 0.f;
    if (t < OUT)
        for (int i = 0; i < IN; ++i) so += v[i] * w[i * OUT + t];
    red[t] = so * so;
    __syncthreads();
    for (int s = 128; s > 0; s >>= 1) { if (t < s) red[t] += red[t + s]; __syncthreads(); }
    if (t == 0) { float ss = red[0]; isig = sqrtf(ss + 1e-12f) / fmaxf(ss, 1e-30f); }
    __syncthreads();

    for (int idx = t; idx < IN * OUT; idx += 256) {
        int i = idx / OUT, j = idx % OUT;
        o[j * IN + i] = (__bf16)(w[idx] * isig);
    }
}

// ---------------------------------------------------------------------------
// Projections: 4 waves / block, 64 pixels / block. The 96KB transposed
// weight blob (theta|phi|g) is async-staged into LDS once per block; every
// B-fragment afterwards is a ds_load_b128.  Dyn LDS = 98304 B.
// ---------------------------------------------------------------------------
__global__ __launch_bounds__(128) void proj_kernel(
    const float* __restrict__ x,
    const __bf16* __restrict__ wt_all,      // [ wt_th | wt_ph | wt_g ] contiguous
    const float* __restrict__ b_th, const float* __restrict__ b_ph,
    const float* __restrict__ b_g,
    __bf16* __restrict__ theta, __bf16* __restrict__ phi_f, __bf16* __restrict__ g_f)
{
    extern __shared__ __align__(64) char smem[];
    __bf16* Wth = (__bf16*)smem;            // 32*256  =  8192 bf16
    __bf16* Wph = Wth + 32 * CIN;           // 32*256
    __bf16* Wg  = Wph + 32 * CIN;           // 128*256 = 32768 bf16

    const int tid  = threadIdx.x;
    const int wv   = tid >> 5;
    const int lane = tid & 31;
    const int row = lane & 15, seg = lane >> 4, col = lane & 15;
    const long pixbase = (long)blockIdx.x * 64 + wv * 16;

    // stage 49152 bf16 = 6144 x 16B units
    __bf16* wl = (__bf16*)smem;
    for (int i = tid; i < 6144; i += 128)
        copy16_g2l(wt_all + (long)i * 8, wl + i * 8);
    async_wait<0>();
    __syncthreads();

    v8f c_th[2] = {{0}}; v8f c_ph[2] = {{0}}; v8f c_g[8] = {{0}};
    for (int k = 0; k < 8; ++k) {                       // 8 x (K=32) steps
        v16bf a;
        const float* xr = x + (pixbase + row) * CIN + k * 32 + seg * 16;
        #pragma unroll
        for (int j = 0; j < 16; ++j) a[j] = (__bf16)xr[j];
        #pragma unroll
        for (int nt = 0; nt < 2; ++nt) {
            v16bf bt = *(const v16bf*)(Wth + (nt * 16 + row) * CIN + k * 32 + seg * 16);
            c_th[nt] = wmma_bf16(a, bt, c_th[nt]);
            v16bf bp = *(const v16bf*)(Wph + (nt * 16 + row) * CIN + k * 32 + seg * 16);
            c_ph[nt] = wmma_bf16(a, bp, c_ph[nt]);
        }
        #pragma unroll
        for (int nt = 0; nt < 8; ++nt) {
            v16bf bg = *(const v16bf*)(Wg + (nt * 16 + row) * CIN + k * 32 + seg * 16);
            c_g[nt] = wmma_bf16(a, bg, c_g[nt]);
        }
    }
    // C/D layout: VGPR r -> matrix row (r + seg*8), column = lane&15
    #pragma unroll
    for (int nt = 0; nt < 2; ++nt) {
        float bth = b_th[nt * 16 + col], bph = b_ph[nt * 16 + col];
        #pragma unroll
        for (int r = 0; r < 8; ++r) {
            long p = pixbase + r + seg * 8;
            theta[p * 32 + nt * 16 + col] = (__bf16)(c_th[nt][r] + bth);
            phi_f[p * 32 + nt * 16 + col] = (__bf16)(c_ph[nt][r] + bph);
        }
    }
    #pragma unroll
    for (int nt = 0; nt < 8; ++nt) {
        float bg = b_g[nt * 16 + col];
        #pragma unroll
        for (int r = 0; r < 8; ++r) {
            long p = pixbase + r + seg * 8;
            g_f[p * 128 + nt * 16 + col] = (__bf16)(c_g[nt][r] + bg);
        }
    }
}

// ---------------------------------------------------------------------------
// 2x2 max pool for phi: [n,64,64,32] -> [n,1024(kp),32]
// ---------------------------------------------------------------------------
__global__ __launch_bounds__(256) void pool_phi_kernel(
    const __bf16* __restrict__ phi_f, __bf16* __restrict__ phi_p)
{
    int idx = blockIdx.x * 256 + threadIdx.x;
    if (idx >= 16 * 1024 * 32) return;
    int ch = idx & 31, kp = (idx >> 5) & 1023, n = idx >> 15;
    int ph = kp >> 5, pw = kp & 31;
    long base = (((long)n * 4096) + ph * 128 + pw * 2) * 32 + ch;
    float m = (float)phi_f[base];
    m = fmaxf(m, (float)phi_f[base + 32]);
    m = fmaxf(m, (float)phi_f[base + 64 * 32]);
    m = fmaxf(m, (float)phi_f[base + 65 * 32]);
    phi_p[idx] = (__bf16)m;
}

// ---------------------------------------------------------------------------
// 2x2 max pool + transpose for g: [n,64,64,128] -> g_t[n,128(ch),1024(kp)]
// ---------------------------------------------------------------------------
__global__ __launch_bounds__(256) void pool_g_kernel(
    const __bf16* __restrict__ g_f, __bf16* __restrict__ g_t)
{
    int idx = blockIdx.x * 256 + threadIdx.x;
    if (idx >= 16 * 1024 * 128) return;
    int ch = idx & 127, kp = (idx >> 7) & 1023, n = idx >> 17;
    int ph = kp >> 5, pw = kp & 31;
    long base = (((long)n * 4096) + ph * 128 + pw * 2) * 128 + ch;
    float m = (float)g_f[base];
    m = fmaxf(m, (float)g_f[base + 128]);
    m = fmaxf(m, (float)g_f[base + 64 * 128]);
    m = fmaxf(m, (float)g_f[base + 65 * 128]);
    g_t[(((long)n * 128) + ch) * 1024 + kp] = (__bf16)m;
}

// ---------------------------------------------------------------------------
// Fused attention: 4 waves / block, one 64-row q-tile per block.
//   phi (64KB) staged to LDS once; g streamed via double-buffered async
//   copies (8KB chunks) overlapped with PV WMMAs; S (128KB) + Ot (16KB) LDS.
// Dyn LDS = 229632 B (~224KB of the 320KB WGP budget).
// ---------------------------------------------------------------------------
__global__ __launch_bounds__(128) void attn_kernel(
    const float* __restrict__ x,
    const __bf16* __restrict__ theta, const __bf16* __restrict__ phi_p,
    const __bf16* __restrict__ g_t,   const __bf16* __restrict__ wt_a,
    const float* __restrict__ b_a,    const float* __restrict__ sigma,
    float* __restrict__ out)
{
    extern __shared__ __align__(64) char smem[];
    __bf16* S      = (__bf16*)smem;                         // 64*1024 = 128KB
    __bf16* PHI    = (__bf16*)(smem + 131072);              // 1024*32 =  64KB
    __bf16* GT     = (__bf16*)(smem + 131072 + 65536);      // 2*128*32 = 16KB
    __bf16* Ot     = (__bf16*)(smem + 131072 + 65536 + 16384);   // 64*128 = 16KB
    float*  rowinv = (float*)(smem + 131072 + 65536 + 16384 + 16384); // 64 f32

    const int tid  = threadIdx.x;
    const int wv   = tid >> 5;
    const int lane = tid & 31;
    const int row = lane & 15, seg = lane >> 4, col = lane & 15;
    const int n  = blockIdx.x >> 6;
    const int qt = blockIdx.x & 63;
    const long qbase = (long)n * 4096 + (long)qt * 64;
    const long qrow0 = qbase + wv * 16;

    // ---- stage full pooled phi for this batch: 4096 x 16B ----
    const __bf16* phin = phi_p + (long)n * 1024 * 32;
    for (int i = tid; i < 4096; i += 128)
        copy16_g2l(phin + (long)i * 8, PHI + i * 8);
    async_wait<0>();
    __syncthreads();

    // ---- S = Q @ Phi^T : 64 WMMAs / wave ----
    v16bf aq = *(const v16bf*)(theta + (qrow0 + row) * 32 + seg * 16);
    __bf16* Sw = S + wv * 16 * 1024;
    for (int kt = 0; kt < 64; ++kt) {
        v16bf b = *(const v16bf*)(PHI + (kt * 16 + row) * 32 + seg * 16);
        v8f c = {0};
        c = wmma_bf16(aq, b, c);
        #pragma unroll
        for (int r = 0; r < 8; ++r)
            Sw[(r + seg * 8) * 1024 + kt * 16 + col] = (__bf16)c[r];
    }

    // ---- softmax over k (row `row`; each half-wave owns 512 columns) ----
    const __bf16* Sr = Sw + row * 1024;
    float m = -1e30f;
    for (int i = seg * 512; i < seg * 512 + 512; ++i)
        m = fmaxf(m, (float)Sr[i]);
    m = fmaxf(m, __shfl_xor(m, 16));
    float sum = 0.f;
    for (int i = seg * 512; i < seg * 512 + 512; ++i) {
        float p = __expf((float)Sr[i] - m);
        sum += p;
        Sw[row * 1024 + i] = (__bf16)p;
    }
    sum += __shfl_xor(sum, 16);
    if (seg == 0) rowinv[wv * 16 + row] = 1.f / sum;

    // ---- O = P @ G : double-buffered g chunks, 256 WMMAs / wave ----
    const __bf16* gtn = g_t + (long)n * 128 * 1024;
    // stage chunk 0 -> buf 0 (512 x 16B units, 4 per thread)
    {
        #pragma unroll
        for (int j = 0; j < 4; ++j) {
            int u = tid + 128 * j;             // 0..511
            int ch = u >> 2, part = u & 3;
            copy16_g2l(gtn + (long)ch * 1024 + 0 * 32 + part * 8,
                       GT + ch * 32 + part * 8);
        }
    }
    v8f o[8] = {{0}};
    for (int kc = 0; kc < 32; ++kc) {
        if (kc + 1 < 32) {
            __bf16* nb = GT + ((kc + 1) & 1) * 4096;
            #pragma unroll
            for (int j = 0; j < 4; ++j) {
                int u = tid + 128 * j;
                int ch = u >> 2, part = u & 3;
                copy16_g2l(gtn + (long)ch * 1024 + (kc + 1) * 32 + part * 8,
                           nb + ch * 32 + part * 8);
            }
            async_wait<4>();                   // previous chunk complete
        } else {
            async_wait<0>();
        }
        __syncthreads();                        // chunk kc visible to all waves
        const __bf16* buf = GT + (kc & 1) * 4096;
        v16bf ap = *(const v16bf*)(Sw + row * 1024 + kc * 32 + seg * 16);
        #pragma unroll
        for (int nt = 0; nt < 8; ++nt) {
            v16bf b = *(const v16bf*)(buf + (nt * 16 + row) * 32 + seg * 16);
            o[nt] = wmma_bf16(ap, b, o[nt]);
        }
        __syncthreads();                        // buffer free for next stage
    }
    #pragma unroll
    for (int nt = 0; nt < 8; ++nt)
        #pragma unroll
        for (int r = 0; r < 8; ++r)
            Ot[(wv * 16 + r + seg * 8) * 128 + nt * 16 + col] =
                (__bf16)(o[nt][r] * rowinv[wv * 16 + r + seg * 8]);
    __syncthreads();

    // ---- out = x + sigma * (O @ Wa + b) : 64 WMMAs / wave ----
    const float sg = sigma[0];
    for (int nt = 0; nt < 16; ++nt) {
        v8f c = {0};
        #pragma unroll
        for (int k = 0; k < 4; ++k) {
            v16bf ao = *(const v16bf*)(Ot + (wv * 16 + row) * 128 + k * 32 + seg * 16);
            v16bf b  = *(const v16bf*)(wt_a + (nt * 16 + row) * 128 + k * 32 + seg * 16);
            c = wmma_bf16(ao, b, c);
        }
        float bias = b_a[nt * 16 + col];
        #pragma unroll
        for (int r = 0; r < 8; ++r) {
            long oidx = (qrow0 + r + seg * 8) * 256 + nt * 16 + col;
            out[oidx] = x[oidx] + sg * (c[r] + bias);
        }
    }
}

// ---------------------------------------------------------------------------
extern "C" void kernel_launch(void* const* d_in, const int* in_sizes, int n_in,
                              void* d_out, int out_size, void* d_ws, size_t ws_size,
                              hipStream_t stream) {
    const float* x       = (const float*)d_in[0];
    const float* w_theta = (const float*)d_in[1];
    const float* b_theta = (const float*)d_in[2];
    const float* u_theta = (const float*)d_in[3];
    const float* w_phi   = (const float*)d_in[4];
    const float* b_phi   = (const float*)d_in[5];
    const float* u_phi   = (const float*)d_in[6];
    const float* w_g     = (const float*)d_in[7];
    const float* b_g     = (const float*)d_in[8];
    const float* u_g     = (const float*)d_in[9];
    const float* w_attn  = (const float*)d_in[10];
    const float* b_attn  = (const float*)d_in[11];
    const float* u_attn  = (const float*)d_in[12];
    const float* sigma   = (const float*)d_in[13];
    float* out = (float*)d_out;

    char* ws = (char*)d_ws;
    __bf16* wt_th = (__bf16*)(ws + 0);               //  32*256*2 = 16 KB
    __bf16* wt_ph = (__bf16*)(ws + (16 << 10));      //  16 KB
    __bf16* wt_g  = (__bf16*)(ws + (32 << 10));      //  64 KB  (blob = 96 KB)
    __bf16* wt_a  = (__bf16*)(ws + (96 << 10));      //  64 KB
    __bf16* theta = (__bf16*)(ws + (160 << 10));               // 4 MB
    __bf16* phi_f = (__bf16*)(ws + (160 << 10) + (4 << 20));   // 4 MB
    __bf16* g_f   = (__bf16*)(ws + (160 << 10) + (8 << 20));   // 16 MB
    __bf16* phi_p = (__bf16*)(ws + (160 << 10) + (24 << 20));  // 1 MB
    __bf16* g_t   = (__bf16*)(ws + (160 << 10) + (25 << 20));  // 4 MB

    sn_kernel<<<4, 256, 0, stream>>>(w_theta, u_theta, wt_th,
                                     w_phi,   u_phi,   wt_ph,
                                     w_g,     u_g,     wt_g,
                                     w_attn,  u_attn,  wt_a);

    proj_kernel<<<1024, 128, 98304, stream>>>(x, wt_th,
                                              b_theta, b_phi, b_g,
                                              theta, phi_f, g_f);

    pool_phi_kernel<<<(16 * 1024 * 32 + 255) / 256, 256, 0, stream>>>(phi_f, phi_p);
    pool_g_kernel<<<(16 * 1024 * 128 + 255) / 256, 256, 0, stream>>>(g_f, g_t);

    attn_kernel<<<1024, 128, 229632, stream>>>(x, theta, phi_p, g_t, wt_a,
                                               b_attn, sigma, out);
}
